// HebbianNetwork_36704790511729
// MI455X (gfx1250) — 16-bit and 8-bit matrix operand order repaired and checked by reading, compiled, not run on hardware
//
#include <hip/hip_runtime.h>
#include <hip/hip_bf16.h>

typedef __bf16 bf16_t;
typedef __attribute__((ext_vector_type(16))) __bf16 v16bf;
typedef __attribute__((ext_vector_type(8)))  float  v8f;
typedef __attribute__((ext_vector_type(8)))  int    v8i;

union FragBF  { uint4 q[2]; v16bf v; };
union FragU8A { uint2 d[4]; v8i  v; };
union FragU8B { uint4 q[2]; v8i  v; };

__device__ __forceinline__ unsigned short f32_to_bf16_rn(float f) {
  unsigned int u = __float_as_uint(f);
  unsigned int r = u + 0x7FFFu + ((u >> 16) & 1u);
  return (unsigned short)(r >> 16);
}
__device__ __forceinline__ float bf16_hi_to_f32(unsigned short h) {
  return __uint_as_float(((unsigned int)h) << 16);
}

__device__ __forceinline__ v8f wmma_bf16(v16bf a, v16bf b, v8f c) {
  return __builtin_amdgcn_wmma_f32_16x16x32_bf16(false, a, false, b, (short)0, c,
                                                 false, false);
}
__device__ __forceinline__ v8i wmma_iu8(v8i a, v8i b, v8i c) {
  return __builtin_amdgcn_wmma_i32_16x16x64_iu8(false, a, false, b, c,
                                                false, false);
}

__device__ __forceinline__ v16bf ld_frag_bf16(const bf16_t* __restrict__ p, int gap) {
  FragBF f;
  f.q[0] = *(const uint4*)(p);
  f.q[1] = *(const uint4*)(p + gap);
  return f.v;
}

__device__ __forceinline__ v8i ld_frag_u8a(const unsigned char* __restrict__ p) {
  FragU8A f;
  f.d[0] = *(const uint2*)(p);
  f.d[1] = *(const uint2*)(p + 8);
  f.d[2] = *(const uint2*)(p + 32);
  f.d[3] = *(const uint2*)(p + 40);
  return f.v;
}
__device__ __forceinline__ v8i ld_frag_u8b(const unsigned char* __restrict__ p) {
  FragU8B f;
  f.q[0] = *(const uint4*)(p);
  f.q[1] = *(const uint4*)(p + 32);
  return f.v;
}


__global__ void split_pad_kernel(const float* __restrict__ in,
                                 unsigned short* __restrict__ hi,
                                 unsigned short* __restrict__ lo,
                                 long rows_in, long rows_out, long cols) {
  long idx = (long)blockIdx.x * 256 + threadIdx.x;
  long total = rows_out * cols;
  if (idx >= total) return;
  long r = idx / cols;
  float v = (r < rows_in) ? in[idx] : 0.0f;
  unsigned short h = f32_to_bf16_rn(v);
  float rest = v - bf16_hi_to_f32(h);
  hi[idx] = h;
  lo[idx] = f32_to_bf16_rn(rest);
}

__global__ void pad_bias_kernel(const float* __restrict__ b, float* __restrict__ bp,
                                int n_in, int n_out) {
  int i = blockIdx.x * 256 + threadIdx.x;
  if (i < n_out) bp[i] = (i < n_in) ? b[i] : 0.0f;
}

__global__ void transpose_thresh_kernel(const float* __restrict__ X,
                                        unsigned char* __restrict__ XT,
                                        long Rows, long Cols) {
  __shared__ float tile[32][33];
  long bx = (long)blockIdx.x * 32;
  long by = (long)blockIdx.y * 32;
  int tx = threadIdx.x, ty = threadIdx.y;
  #pragma unroll
  for (int r = ty; r < 32; r += 8)
    tile[r][tx] = X[(by + r) * Cols + bx + tx];
  __syncthreads();
  #pragma unroll
  for (int c = ty; c < 32; c += 8)
    XT[(bx + c) * Rows + by + tx] = (tile[tx][c] > 0.0f) ? 1u : 0u;
}

template <bool RELU_SPLIT>
__global__ __launch_bounds__(32)
void gemm_layer_kernel(const bf16_t* __restrict__ Ah, const bf16_t* __restrict__ Al,
                       const bf16_t* __restrict__ Wh, const bf16_t* __restrict__ Wl,
                       const float* __restrict__ bias,
                       unsigned short* __restrict__ outH,
                       unsigned short* __restrict__ outL,
                       unsigned char* __restrict__ actT,
                       float* __restrict__ zOut, int Nout,
                       int M, int N, int K) {
  const int lane = threadIdx.x;
  const int nl = lane & 15, half = lane >> 4;
  const int m0 = blockIdx.y * 32;
  const int n0 = blockIdx.x * 64;

  const bf16_t* pah[2];
  const bf16_t* pal[2];
  const bf16_t* pwh[4];
  const bf16_t* pwl[4];
  #pragma unroll
  for (int i = 0; i < 2; ++i) {
    const int row = m0 + i * 16 + nl;
    pah[i] = Ah + (long)row * K + half * 8;
    pal[i] = Al + (long)row * K + half * 8;
  }
  #pragma unroll
  for (int j = 0; j < 4; ++j) {
    const int wrow = n0 + j * 16 + nl;
    pwh[j] = Wh + (long)wrow * K + half * 8;
    pwl[j] = Wl + (long)wrow * K + half * 8;
  }

  v8f acc[2][4] = {};

  for (int k0 = 0; k0 < K; k0 += 32) {
    v16bf ah[2], al[2], bh[4], bl[4];
    #pragma unroll
    for (int i = 0; i < 2; ++i) {
      ah[i] = ld_frag_bf16(pah[i] + k0, 16);
      al[i] = ld_frag_bf16(pal[i] + k0, 16);
      __builtin_prefetch((const void*)(pah[i] + k0 + 32), 0, 1);
    }
    #pragma unroll
    for (int j = 0; j < 4; ++j) {
      bh[j] = ld_frag_bf16(pwh[j] + k0, 16);
      bl[j] = ld_frag_bf16(pwl[j] + k0, 16);
      __builtin_prefetch((const void*)(pwh[j] + k0 + 32), 0, 1);
    }
    #pragma unroll
    for (int i = 0; i < 2; ++i)
      #pragma unroll
      for (int j = 0; j < 4; ++j) {
        acc[i][j] = wmma_bf16(ah[i], bh[j], acc[i][j]);
        acc[i][j] = wmma_bf16(ah[i], bl[j], acc[i][j]);
        acc[i][j] = wmma_bf16(al[i], bh[j], acc[i][j]);
      }
  }

  #pragma unroll
  for (int i = 0; i < 2; ++i) {
    #pragma unroll
    for (int j = 0; j < 4; ++j) {
      const int n = n0 + j * 16 + nl;
      const float bv = bias[n];
      #pragma unroll
      for (int r = 0; r < 8; ++r) {
        const int m = m0 + i * 16 + half * 8 + r;
        const float z = acc[i][j][r] + bv;
        actT[(long)n * M + m] = (z > 0.0f) ? 1u : 0u;
        if constexpr (RELU_SPLIT) {
          const float a = (z > 0.0f) ? z : 0.0f;
          const unsigned short h = f32_to_bf16_rn(a);
          const float rest = a - bf16_hi_to_f32(h);
          outH[(long)m * N + n] = h;
          outL[(long)m * N + n] = f32_to_bf16_rn(rest);
        } else {
          if (n < Nout) zOut[(long)m * Nout + n] = z;
        }
      }
    }
  }
}

struct CoFrags { v8i a[2]; v8i b[4]; };

__device__ __forceinline__ void ld_cofrags(CoFrags& F,
                                           const unsigned char* const* pu,
                                           const unsigned char* const* pv,
                                           int co) {
  #pragma unroll
  for (int i = 0; i < 2; ++i) F.a[i] = ld_frag_u8a(pu[i] + co);
  #pragma unroll
  for (int j = 0; j < 4; ++j) F.b[j] = ld_frag_u8b(pv[j] + co);
}

__global__ __launch_bounds__(32)
void coact_gemm_kernel(const unsigned char* __restrict__ UT,
                       const unsigned char* __restrict__ VT,
                       float* __restrict__ out, int Bn, int Jout) {
  const int lane = threadIdx.x;
  const int nl = lane & 15, half = lane >> 4;
  const int i0 = blockIdx.y * 32;
  const int j0 = blockIdx.x * 64;

  const unsigned char* pu[2];
  const unsigned char* pv[4];
  #pragma unroll
  for (int i = 0; i < 2; ++i) pu[i] = UT + (long)(i0 + i * 16 + nl) * Bn + half * 16;
  #pragma unroll
  for (int j = 0; j < 4; ++j) pv[j] = VT + (long)(j0 + j * 16 + nl) * Bn + half * 16;

  v8i acc[2][4] = {};
  CoFrags F0, F1;

  ld_cofrags(F0, pu, pv, 0);
  const int kc = Bn / 64;
  for (int c = 0; c < kc; c += 2) {
    const int co = c * 64;
    ld_cofrags(F1, pu, pv, co + 64);
    __builtin_prefetch((const void*)(pu[0] + co + 128), 0, 1);
    __builtin_prefetch((const void*)(pv[0] + co + 128), 0, 1);
    #pragma unroll
    for (int i = 0; i < 2; ++i)
      #pragma unroll
      for (int j = 0; j < 4; ++j)
        acc[i][j] = wmma_iu8(F0.a[i], F0.b[j], acc[i][j]);
    ld_cofrags(F0, pu, pv, co + 128);
    #pragma unroll
    for (int i = 0; i < 2; ++i)
      #pragma unroll
      for (int j = 0; j < 4; ++j)
        acc[i][j] = wmma_iu8(F1.a[i], F1.b[j], acc[i][j]);
  }

  #pragma unroll
  for (int i = 0; i < 2; ++i)
    #pragma unroll
    for (int j = 0; j < 4; ++j) {
      const int jj = j0 + j * 16 + nl;
      if (jj < Jout) {
        #pragma unroll
        for (int r = 0; r < 8; ++r) {
          const int ii = i0 + i * 16 + half * 8 + r;
          out[(long)ii * Jout + jj] = (float)acc[i][j][r];
        }
      }
    }
}

namespace cfg {
constexpr long D_IN = 1024, H1 = 2048, H2 = 2048, NC = 1000, NCP = 1024, B = 8192;

constexpr size_t SZ_X   = (size_t)B * D_IN * 2;
constexpr size_t SZ_W1  = (size_t)H1 * D_IN * 2;
constexpr size_t SZ_W2  = (size_t)H2 * H1 * 2;
constexpr size_t SZ_W3P = (size_t)NCP * H2 * 2;
constexpr size_t SZ_A   = (size_t)B * H1 * 2;
constexpr size_t SZ_PT  = (size_t)D_IN * B;
constexpr size_t SZ_AT  = (size_t)H1 * B;
constexpr size_t SZ_A3T = (size_t)NCP * B;

constexpr size_t OFF_XH   = 0;
constexpr size_t OFF_XL   = OFF_XH + SZ_X;
constexpr size_t OFF_W1H  = OFF_XL + SZ_X;
constexpr size_t OFF_W1L  = OFF_W1H + SZ_W1;
constexpr size_t OFF_W2H  = OFF_W1L + SZ_W1;
constexpr size_t OFF_W2L  = OFF_W2H + SZ_W2;
constexpr size_t OFF_W3H  = OFF_W2L + SZ_W2;
constexpr size_t OFF_W3L  = OFF_W3H + SZ_W3P;
constexpr size_t OFF_B3P  = OFF_W3L + SZ_W3P;
constexpr size_t OFF_A1H  = OFF_B3P + 4096;
constexpr size_t OFF_A1L  = OFF_A1H + SZ_A;
constexpr size_t OFF_A2H  = OFF_A1L + SZ_A;
constexpr size_t OFF_A2L  = OFF_A2H + SZ_A;
constexpr size_t OFF_PT   = OFF_A2L + SZ_A;
constexpr size_t OFF_A1T  = OFF_PT + SZ_PT;
constexpr size_t OFF_A2T  = OFF_A1T + SZ_AT;
constexpr size_t OFF_A3T  = OFF_A2T + SZ_AT;
constexpr size_t WS_NEEDED = OFF_A3T + SZ_A3T + 4096;

constexpr size_t OUT_LOGITS = 0;
constexpr size_t OUT_C0 = (size_t)B * NC;
constexpr size_t OUT_C1 = OUT_C0 + (size_t)D_IN * H1;
constexpr size_t OUT_C2 = OUT_C1 + (size_t)H1 * H2;
}

extern "C" void kernel_launch(void* const* d_in, const int* in_sizes, int n_in,
                              void* d_out, int out_size, void* d_ws, size_t ws_size,
                              hipStream_t stream) {
  using namespace cfg;
  if (ws_size < WS_NEEDED) return;

  const float* x  = (const float*)d_in[0];
  const float* W1 = (const float*)d_in[1];
  const float* b1 = (const float*)d_in[2];
  const float* W2 = (const float*)d_in[3];
  const float* b2 = (const float*)d_in[4];
  const float* W3 = (const float*)d_in[5];
  const float* b3 = (const float*)d_in[6];
  float* out = (float*)d_out;
  char* ws = (char*)d_ws;

  bf16_t* xh  = (bf16_t*)(ws + OFF_XH);
  bf16_t* xl  = (bf16_t*)(ws + OFF_XL);
  bf16_t* w1h = (bf16_t*)(ws + OFF_W1H);
  bf16_t* w1l = (bf16_t*)(ws + OFF_W1L);
  bf16_t* w2h = (bf16_t*)(ws + OFF_W2H);
  bf16_t* w2l = (bf16_t*)(ws + OFF_W2L);
  bf16_t* w3h = (bf16_t*)(ws + OFF_W3H);
  bf16_t* w3l = (bf16_t*)(ws + OFF_W3L);
  float*  b3p = (float*)(ws + OFF_B3P);
  bf16_t* a1h = (bf16_t*)(ws + OFF_A1H);
  bf16_t* a1l = (bf16_t*)(ws + OFF_A1L);
  bf16_t* a2h = (bf16_t*)(ws + OFF_A2H);
  bf16_t* a2l = (bf16_t*)(ws + OFF_A2L);
  unsigned char* pT  = (unsigned char*)(ws + OFF_PT);
  unsigned char* a1T = (unsigned char*)(ws + OFF_A1T);
  unsigned char* a2T = (unsigned char*)(ws + OFF_A2T);
  unsigned char* a3T = (unsigned char*)(ws + OFF_A3T);

  auto blocks1d = [](long total) { return (unsigned)((total + 255) / 256); };

  split_pad_kernel<<<blocks1d(B * D_IN), 256, 0, stream>>>(
      x, (unsigned short*)xh, (unsigned short*)xl, B, B, D_IN);
  split_pad_kernel<<<blocks1d(H1 * D_IN), 256, 0, stream>>>(
      W1, (unsigned short*)w1h, (unsigned short*)w1l, H1, H1, D_IN);
  split_pad_kernel<<<blocks1d(H2 * H1), 256, 0, stream>>>(
      W2, (unsigned short*)w2h, (unsigned short*)w2l, H2, H2, H1);
  split_pad_kernel<<<blocks1d(NCP * H2), 256, 0, stream>>>(
      W3, (unsigned short*)w3h, (unsigned short*)w3l, NC, NCP, H2);
  pad_bias_kernel<<<blocks1d(NCP), 256, 0, stream>>>(b3, b3p, (int)NC, (int)NCP);

  {
    dim3 grid((unsigned)(D_IN / 32), (unsigned)(B / 32));
    transpose_thresh_kernel<<<grid, dim3(32, 8), 0, stream>>>(x, pT, B, D_IN);
  }

  {
    dim3 grid((unsigned)(H1 / 64), (unsigned)(B / 32));
    gemm_layer_kernel<true><<<grid, 32, 0, stream>>>(
        xh, xl, w1h, w1l, b1, (unsigned short*)a1h, (unsigned short*)a1l,
        a1T, nullptr, 0, (int)B, (int)H1, (int)D_IN);
  }
  {
    dim3 grid((unsigned)(H2 / 64), (unsigned)(B / 32));
    gemm_layer_kernel<true><<<grid, 32, 0, stream>>>(
        a1h, a1l, w2h, w2l, b2, (unsigned short*)a2h, (unsigned short*)a2l,
        a2T, nullptr, 0, (int)B, (int)H2, (int)H1);
  }
  {
    dim3 grid((unsigned)(NCP / 64), (unsigned)(B / 32));
    gemm_layer_kernel<false><<<grid, 32, 0, stream>>>(
        a2h, a2l, w3h, w3l, b3p, nullptr, nullptr,
        a3T, out + OUT_LOGITS, (int)NC, (int)B, (int)NCP, (int)H2);
  }

  {
    dim3 grid((unsigned)(H1 / 64), (unsigned)(D_IN / 32));
    coact_gemm_kernel<<<grid, 32, 0, stream>>>(pT, a1T, out + OUT_C0, (int)B, (int)H1);
  }
  {
    dim3 grid((unsigned)(H2 / 64), (unsigned)(H1 / 32));
    coact_gemm_kernel<<<grid, 32, 0, stream>>>(a1T, a2T, out + OUT_C1, (int)B, (int)H2);
  }
  {
    dim3 grid((unsigned)(NCP / 64), (unsigned)(H2 / 32));
    coact_gemm_kernel<<<grid, 32, 0, stream>>>(a2T, a3T, out + OUT_C2, (int)B, (int)NC);
  }
}
